// PicoChatGPT_84859963834601
// MI455X (gfx1250) — matrix operand order repaired and checked by reading, not compiled, not run on hardware
//
#include <hip/hip_runtime.h>
#include <hip/hip_bf16.h>
#include <math.h>

// ---------------------------------------------------------------------------
// Types for CDNA5 WMMA (wave32): V_WMMA_F32_16X16X32_BF16
// ---------------------------------------------------------------------------
typedef __bf16         v16bf __attribute__((ext_vector_type(16)));
typedef float          v8f   __attribute__((ext_vector_type(8)));
typedef unsigned int   v4u   __attribute__((ext_vector_type(4)));
typedef int            v8i   __attribute__((ext_vector_type(8)));
typedef int            v4i   __attribute__((ext_vector_type(4)));
typedef unsigned short us8   __attribute__((ext_vector_type(8)));

__device__ __forceinline__ __bf16 us2bf(unsigned short s) {
    return __builtin_bit_cast(__bf16, s);
}

__device__ __forceinline__ unsigned short f32_to_bf16(float f) {
    unsigned int u = __float_as_uint(f);
    unsigned int rounding = 0x7FFFu + ((u >> 16) & 1u);
    return (unsigned short)((u + rounding) >> 16);
}

__device__ __forceinline__ __bf16 f2bf(float f) {
    return __builtin_bit_cast(__bf16, f32_to_bf16(f));
}

// ---------------------------------------------------------------------------
// fp32 -> bf16 conversion (weights once per launch, activations per use)
// ---------------------------------------------------------------------------
__global__ __launch_bounds__(256) void cvt_bf16_kernel(const float* __restrict__ src,
                                                       unsigned short* __restrict__ dst,
                                                       size_t n) {
    for (size_t i = (size_t)blockIdx.x * 256 + threadIdx.x; i < n;
         i += (size_t)gridDim.x * 256)
        dst[i] = f32_to_bf16(src[i]);
}

__global__ __launch_bounds__(256) void relu2_bf16_kernel(const float* __restrict__ src,
                                                         unsigned short* __restrict__ dst,
                                                         size_t n) {
    for (size_t i = (size_t)blockIdx.x * 256 + threadIdx.x; i < n;
         i += (size_t)gridDim.x * 256) {
        float v = fmaxf(src[i], 0.0f);
        dst[i] = f32_to_bf16(v * v);
    }
}

// ---------------------------------------------------------------------------
// WMMA GEMM: C[M,N] = A[M,K] @ B[K,N], bf16 in / f32 out.
// Block: 256 threads = 8 wave32.  Block tile 64x128, K step 32.
// Double-buffered LDS pipeline: TDM for A (TENSORcnt), wide global loads +
// fragment-major repack for B.
// epilogue: 0 = store, 1 = resid[o] + acc, 2 = CAP*tanh(acc/CAP), col < Nout
// ---------------------------------------------------------------------------
#define GBM 64
#define GBN 128
#define GBK 32

__global__ __launch_bounds__(256) void gemm_bf16_wmma(
    const unsigned short* __restrict__ A,
    const unsigned short* __restrict__ B,
    float* __restrict__ C,
    const float* __restrict__ resid,
    int M, int N, int K, int ldc, int Nout, int epilogue)
{
    __shared__ unsigned short As[2][GBM][GBK];                                   // 2 x 4 KB (TDM dest)
    __shared__ unsigned short Bp[2][GBN / 16][32][16] __attribute__((aligned(32))); // 2 x 8 KB

    const int tid   = threadIdx.x;
    const int lane  = tid & 31;
    const int wid   = tid >> 5;       // 0..7
    const int wm    = wid >> 2;       // 0..1 -> row block of 32
    const int wn    = wid & 3;        // 0..3 -> col block of 32
    const int half  = lane >> 4;      // 0/1
    const int l15   = lane & 15;

    const int tile_m0 = blockIdx.y * GBM;
    const int tile_n0 = blockIdx.x * GBN;

    auto stage_tile = [&](int buf, int k0) {
        if (tid < 32) {
            unsigned long long ga =
                (unsigned long long)(uintptr_t)(A + (size_t)tile_m0 * K + k0);
            unsigned ldsoff = (unsigned)(uintptr_t)(&As[buf][0][0]);
            v4u g0 = { 1u,
                       ldsoff,
                       (unsigned)(ga & 0xFFFFFFFFu),
                       (unsigned)((ga >> 32) & 0x01FFFFFFu) | 0x80000000u };
            v8i g1 = { (int)0x00010000u,
                       (int)(((unsigned)K & 0xFFFFu) << 16),
                       (int)((((unsigned)K >> 16) & 0xFFFFu) | (((unsigned)M & 0xFFFFu) << 16)),
                       (int)((((unsigned)M >> 16) & 0xFFFFu) | ((unsigned)GBK << 16)),
                       (int)GBM,
                       (int)K,
                       0, 0 };
            v4i z4 = { 0, 0, 0, 0 };
            v8i z8 = { 0, 0, 0, 0, 0, 0, 0, 0 };
            __builtin_amdgcn_tensor_load_to_lds(g0, g1, z4, z4, z8, 0);
        }
        int r2 = (tid >> 4) << 1;     // 0,2,..,30
        int c8 = (tid & 15) << 3;     // 0,8,..,120
        const us8 v0 = *reinterpret_cast<const us8*>(
            &B[(size_t)(k0 + r2) * N + tile_n0 + c8]);
        const us8 v1 = *reinterpret_cast<const us8*>(
            &B[(size_t)(k0 + r2 + 1) * N + tile_n0 + c8]);
        int g     = c8 >> 4;
        int lbase = ((r2 >> 4) << 4) | (c8 & 15);
        int eidx  = r2 & 15;
        #pragma unroll
        for (int e = 0; e < 8; ++e) {
            unsigned w = (unsigned)v0[e] | ((unsigned)v1[e] << 16);
            *reinterpret_cast<unsigned*>(&Bp[buf][g][lbase + e][eidx]) = w;
        }
    };

    v8f acc[2][2] = {};

    stage_tile(0, 0);
    if (tid < 32) __builtin_amdgcn_s_wait_tensorcnt(0);
    __syncthreads();

    for (int k0 = 0; k0 < K; k0 += GBK) {
        int cur = (k0 >> 5) & 1;
        if (k0 + GBK < K) stage_tile(cur ^ 1, k0 + GBK);
        if (k0 + 2 * GBK < K) {
            __builtin_prefetch(&B[(size_t)(k0 + 2 * GBK + (tid >> 7)) * N + tile_n0 + (tid & 127)], 0, 0);
        }

        v16bf bfrag[2];
        #pragma unroll
        for (int ni = 0; ni < 2; ++ni) {
            bfrag[ni] = *reinterpret_cast<const v16bf*>(&Bp[cur][wn * 2 + ni][lane][0]);
        }
        #pragma unroll
        for (int mi = 0; mi < 2; ++mi) {
            int m = wm * 32 + mi * 16 + l15;
            v16bf af;
            #pragma unroll
            for (int e = 0; e < 8; ++e) {
                int kk = half * 16 + 2 * e;
                af[2 * e]     = us2bf(As[cur][m][kk]);
                af[2 * e + 1] = us2bf(As[cur][m][kk + 1]);
            }
            #pragma unroll
            for (int ni = 0; ni < 2; ++ni) {
                acc[mi][ni] = __builtin_amdgcn_wmma_f32_16x16x32_bf16(
                    false, af, false, bfrag[ni], (short)0, acc[mi][ni], false, false);
            }
        }

        if (tid < 32) __builtin_amdgcn_s_wait_tensorcnt(0);
        __syncthreads();
    }

    #pragma unroll
    for (int mi = 0; mi < 2; ++mi) {
        #pragma unroll
        for (int ni = 0; ni < 2; ++ni) {
            int gn     = tile_n0 + wn * 32 + ni * 16 + l15;
            int gmbase = tile_m0 + wm * 32 + mi * 16 + half * 8;
            #pragma unroll
            for (int r = 0; r < 8; ++r) {
                int gm = gmbase + r;
                float val = acc[mi][ni][r];
                size_t o = (size_t)gm * ldc + gn;
                if (epilogue == 0) {
                    C[o] = val;
                } else if (epilogue == 1) {
                    C[o] = resid[o] + val;
                } else {
                    if (gn < Nout) C[o] = 15.0f * tanhf(val * (1.0f / 15.0f));
                }
            }
        }
    }
}

// ---------------------------------------------------------------------------
// Fused embed + rmsnorm:  x = rmsnorm(wte[idx]),  x0 = x
// ---------------------------------------------------------------------------
__global__ __launch_bounds__(256) void embed_kernel(const int* __restrict__ idx,
                                                    const float* __restrict__ wte,
                                                    float* __restrict__ x,
                                                    float* __restrict__ x0,
                                                    int D)
{
    __shared__ float red[256];
    int row = blockIdx.x;
    const float* w = wte + (size_t)idx[row] * D;
    float ssq = 0.f;
    for (int d = threadIdx.x; d < D; d += 256) { float v = w[d]; ssq += v * v; }
    red[threadIdx.x] = ssq; __syncthreads();
    for (int s = 128; s > 0; s >>= 1) {
        if (threadIdx.x < s) red[threadIdx.x] += red[threadIdx.x + s];
        __syncthreads();
    }
    float r = rsqrtf(red[0] / (float)D + 1e-6f);
    for (int d = threadIdx.x; d < D; d += 256) {
        float v = w[d] * r;
        x [(size_t)row * D + d] = v;
        x0[(size_t)row * D + d] = v;
    }
}

// x = rl[l]*x + xl[l]*x0
__global__ __launch_bounds__(256) void resid_kernel(float* __restrict__ x,
                                                    const float* __restrict__ x0,
                                                    const float* __restrict__ rl,
                                                    const float* __restrict__ xl,
                                                    int layer, size_t n)
{
    float a = rl[layer], b = xl[layer];
    for (size_t i = (size_t)blockIdx.x * 256 + threadIdx.x; i < n;
         i += (size_t)gridDim.x * 256)
        x[i] = a * x[i] + b * x0[i];
}

// h = rmsnorm(x) -> bf16 (GEMM A operand), optionally also fp32 copy
__global__ __launch_bounds__(256) void rmsnorm_out_kernel(const float* __restrict__ x,
                                                          unsigned short* __restrict__ hb,
                                                          float* __restrict__ hf,
                                                          int D)
{
    __shared__ float red[256];
    int row = blockIdx.x;
    const float* xr = x + (size_t)row * D;
    float ssq = 0.f;
    for (int d = threadIdx.x; d < D; d += 256) { float v = xr[d]; ssq += v * v; }
    red[threadIdx.x] = ssq; __syncthreads();
    for (int s = 128; s > 0; s >>= 1) {
        if (threadIdx.x < s) red[threadIdx.x] += red[threadIdx.x + s];
        __syncthreads();
    }
    float r = rsqrtf(red[0] / (float)D + 1e-6f);
    for (int d = threadIdx.x; d < D; d += 256) {
        float v = xr[d] * r;
        hb[(size_t)row * D + d] = f32_to_bf16(v);
        if (hf) hf[(size_t)row * D + d] = v;
    }
}

// ---------------------------------------------------------------------------
// RoPE + rmsnorm per head row, writing bf16: one wave32 per (token, head);
// lane owns dims (lane, lane+32).  in: f32 [BT, H, 64] -> out: bf16 same.
// ---------------------------------------------------------------------------
__global__ __launch_bounds__(256) void rope_rms_kernel(const float* __restrict__ xin,
                                                       unsigned short* __restrict__ xout,
                                                       int BT, int T, int H)
{
    int wid  = blockIdx.x * 8 + (threadIdx.x >> 5);
    int lane = threadIdx.x & 31;
    if (wid >= BT * H) return;
    int h  = wid % H;
    int bt = wid / H;
    int t  = bt % T;

    const float* row = xin + ((size_t)bt * H + h) * 64;
    float x1 = row[lane], x2 = row[lane + 32];
    float inv_freq = __powf(10000.0f, -(float)lane / 64.0f);
    float ang = (float)t * inv_freq;
    float s, c;
    __sincosf(ang, &s, &c);
    float o1 =  x1 * c + x2 * s;
    float o2 = -x1 * s + x2 * c;

    float ssq = o1 * o1 + o2 * o2;
    ssq += __shfl_xor(ssq, 16, 32);
    ssq += __shfl_xor(ssq, 8, 32);
    ssq += __shfl_xor(ssq, 4, 32);
    ssq += __shfl_xor(ssq, 2, 32);
    ssq += __shfl_xor(ssq, 1, 32);
    float r = rsqrtf(ssq / 64.0f + 1e-6f);
    unsigned short* orow = xout + ((size_t)bt * H + h) * 64;
    orow[lane]      = f32_to_bf16(o1 * r);
    orow[lane + 32] = f32_to_bf16(o2 * r);
}

// ---------------------------------------------------------------------------
// Value-embed gating:  v += 2*sigmoid(h[:, :32] @ gw[j]) * ve_emb[j][idx]
// ---------------------------------------------------------------------------
__global__ __launch_bounds__(256) void ve_kernel(const float* __restrict__ hf,
                                                 const int* __restrict__ idx,
                                                 const float* __restrict__ gw,
                                                 const float* __restrict__ ve,
                                                 float* __restrict__ v,
                                                 int j, int BT)
{
    int i = blockIdx.x * 256 + threadIdx.x;
    if (i >= BT * 4) return;
    int kvh = i & 3;
    int bt  = i >> 2;
    const float* g = gw + (size_t)j * 32 * 4;
    float acc = 0.f;
    #pragma unroll
    for (int c = 0; c < 32; ++c) acc += hf[(size_t)bt * 768 + c] * g[c * 4 + kvh];
    float gate = 2.0f / (1.0f + __expf(-acc));
    const float* vr = ve + ((size_t)j * 32768 + idx[bt]) * 256 + kvh * 64;
    float* vd = v + (size_t)bt * 256 + kvh * 64;
    #pragma unroll
    for (int d = 0; d < 64; ++d) vd[d] += gate * vr[d];
}

// V transpose + bf16:  vt[b][kvh][d][t] = bf16(v[b][t][kvh][d])
__global__ __launch_bounds__(256) void vt_bf16_kernel(const float* __restrict__ v,
                                                      unsigned short* __restrict__ vt,
                                                      int T, int BT)
{
    for (int i = blockIdx.x * 256 + threadIdx.x; i < BT * 256;
         i += gridDim.x * 256) {
        int d   = i & 63;
        int kvh = (i >> 6) & 3;
        int bt  = i >> 8;
        int b = bt / T, t = bt % T;
        vt[(((size_t)b * 4 + kvh) * 64 + d) * T + t] = f32_to_bf16(v[i]);
    }
}

// ---------------------------------------------------------------------------
// WMMA flash attention.  One wave32 per (b, h, 16-query tile).
// S^T = K.Q^T so softmax rows live per-lane (8 regs + 1 shfl_xor(16)), and
// P^T registers map directly onto the A fragment of P.V (no lane shuffles).
// Keys processed in 32-blocks: 4 WMMA (S^T) + 4 WMMA (P.V).
// ---------------------------------------------------------------------------
__global__ __launch_bounds__(256) void attn_wmma_kernel(
    const unsigned short* __restrict__ qbh, // [B*T, 12, 64] bf16 (rope+rms)
    const unsigned short* __restrict__ kbh, // [B*T, 4, 64]  bf16 (rope+rms)
    const unsigned short* __restrict__ vtb, // [B, 4, 64, T] bf16 (transposed)
    float* __restrict__ y,                  // [B*T, 12, 64] f32
    int T, int BT, int W)
{
    const int gw   = blockIdx.x * 8 + (threadIdx.x >> 5);
    const int lane = threadIdx.x & 31;
    const int half = lane >> 4;
    const int l15  = lane & 15;
    const int QT   = T >> 4;
    if (gw >= (BT / 16) * 12) return;

    int qt  = gw % QT;
    int rem = gw / QT;
    int h   = rem % 12;
    int b   = rem / 12;
    int kvh = h / 3;
    int q0  = qt << 4;

    // Q^T B-fragments (lane = query column, feats 8*half+0..7 then 16+8*half+0..7 of each 32), reused all loop.
    const unsigned short* qbase = qbh + (((size_t)b * T + q0) * 12 + h) * 64;
    v16bf qtf[2];
    #pragma unroll
    for (int s = 0; s < 2; ++s) {
        const us8* p = reinterpret_cast<const us8*>(
            qbase + (size_t)l15 * 12 * 64 + s * 32 + half * 8);
        us8 r0 = p[0], r1 = p[2];
        v16bf f;
        #pragma unroll
        for (int e = 0; e < 8; ++e) { f[e] = us2bf(r0[e]); f[8 + e] = us2bf(r1[e]); }
        qtf[s] = f;
    }

    const int tq = q0 + l15;                   // this lane's query position
    float mrun = -1e30f, lrun = 0.f;
    v8f vo[4] = {};

    int kstart = q0 - W; if (kstart < 0) kstart = 0;
    kstart &= ~31;
    const int kend = q0 + 15;

    for (int kb = kstart; kb <= kend; kb += 32) {
        // ---- S^T = K . Q^T for two 16-key tiles ----
        v8f st0 = {}, st1 = {};
        #pragma unroll
        for (int s = 0; s < 2; ++s) {
            const unsigned short* k0p =
                kbh + (((size_t)b * T + kb + l15) * 4 + kvh) * 64 + s * 32 + half * 8;
            const unsigned short* k1p =
                kbh + (((size_t)b * T + kb + 16 + l15) * 4 + kvh) * 64 + s * 32 + half * 8;
            us8 a0 = *reinterpret_cast<const us8*>(k0p);
            us8 a1 = *reinterpret_cast<const us8*>(k0p + 16);
            us8 b0 = *reinterpret_cast<const us8*>(k1p);
            us8 b1 = *reinterpret_cast<const us8*>(k1p + 16);
            v16bf kf0, kf1;
            #pragma unroll
            for (int e = 0; e < 8; ++e) {
                kf0[e] = us2bf(a0[e]); kf0[8 + e] = us2bf(a1[e]);
                kf1[e] = us2bf(b0[e]); kf1[8 + e] = us2bf(b1[e]);
            }
            st0 = __builtin_amdgcn_wmma_f32_16x16x32_bf16(false, kf0, false, qtf[s], (short)0, st0, false, false);
            st1 = __builtin_amdgcn_wmma_f32_16x16x32_bf16(false, kf1, false, qtf[s], (short)0, st1, false, false);
        }

        // ---- online softmax over the 32 keys (per-query = per lane column) --
        float m_old = mrun;
        float sc0[8], sc1[8];
        float tmax0 = -1e30f, tmax1 = -1e30f;
        #pragma unroll
        for (int r = 0; r < 8; ++r) {
            int tk0 = kb + half * 8 + r;
            int tk1 = kb + 16 + half * 8 + r;
            float s0 = st0[r] * 0.125f;
            float s1 = st1[r] * 0.125f;
            sc0[r] = (tk0 <= tq && tq - tk0 <= W) ? s0 : -1e30f;
            sc1[r] = (tk1 <= tq && tq - tk1 <= W) ? s1 : -1e30f;
            tmax0 = fmaxf(tmax0, sc0[r]);
            tmax1 = fmaxf(tmax1, sc1[r]);
        }
        tmax0 = fmaxf(tmax0, __shfl_xor(tmax0, 16, 32));
        tmax1 = fmaxf(tmax1, __shfl_xor(tmax1, 16, 32));
        float mn0 = fmaxf(mrun, tmax0);
        float mn1 = fmaxf(mn0, tmax1);
        float c01 = __expf(mn0 - mn1);       // rescale tile0 -> tile1 scale

        float p0[8], p1[8];
        float s0sum = 0.f, s1sum = 0.f;
        #pragma unroll
        for (int r = 0; r < 8; ++r) {
            p0[r] = (sc0[r] < -1e29f) ? 0.f : __expf(sc0[r] - mn0);
            p1[r] = (sc1[r] < -1e29f) ? 0.f : __expf(sc1[r] - mn1);
            p0[r] *= c01;
            s0sum += p0[r];
            s1sum += p1[r];
        }
        float tsum = s0sum + s1sum;
        tsum += __shfl_xor(tsum, 16, 32);
        float bcorr = __expf(m_old - mn1);   // total rescale for O, l
        lrun = lrun * bcorr + tsum;
        mrun = mn1;

        // ---- rescale O accumulators (rows = queries: need bcorr[q] per row) --
        float oc[8];
        #pragma unroll
        for (int r = 0; r < 8; ++r) oc[r] = __shfl(bcorr, half * 8 + r, 32);
        #pragma unroll
        for (int g = 0; g < 4; ++g)
            #pragma unroll
            for (int r = 0; r < 8; ++r) vo[g][r] *= oc[r];

        // ---- P fragment (A-layout, lane = query row; halves pick key groups) -
        v16bf pf;
        #pragma unroll
        for (int e = 0; e < 8; ++e) { pf[e] = f2bf(p0[e]); pf[8 + e] = f2bf(p1[e]); }

        // ---- P . V over 4 dim-groups ----
        #pragma unroll
        for (int g = 0; g < 4; ++g) {
            const unsigned short* vr =
                vtb + (((size_t)b * 4 + kvh) * 64 + g * 16 + l15) * T + kb + half * 8;
            us8 r0 = *reinterpret_cast<const us8*>(vr);
            us8 r1 = *reinterpret_cast<const us8*>(vr + 16);
            v16bf vf;
            #pragma unroll
            for (int e = 0; e < 8; ++e) { vf[e] = us2bf(r0[e]); vf[8 + e] = us2bf(r1[e]); }
            vo[g] = __builtin_amdgcn_wmma_f32_16x16x32_bf16(false, pf, false, vf, (short)0, vo[g], false, false);
        }
    }

    // ---- normalize + store (O rows = queries half*8+r, cols = dims) ----
    float inv = 1.0f / lrun;
    float oinv[8];
    #pragma unroll
    for (int r = 0; r < 8; ++r) oinv[r] = __shfl(inv, half * 8 + r, 32);
    #pragma unroll
    for (int g = 0; g < 4; ++g) {
        #pragma unroll
        for (int r = 0; r < 8; ++r) {
            int tqm = q0 + half * 8 + r;
            y[(((size_t)b * T + tqm) * 12 + h) * 64 + g * 16 + l15] = vo[g][r] * oinv[r];
        }
    }
}

// ---------------------------------------------------------------------------
// Launcher
// ---------------------------------------------------------------------------
extern "C" void kernel_launch(void* const* d_in, const int* in_sizes, int n_in,
                              void* d_out, int out_size, void* d_ws, size_t ws_size,
                              hipStream_t stream) {
    const int*   idx  = (const int*)d_in[0];
    const float* wte  = (const float*)d_in[1];
    const float* lmw  = (const float*)d_in[2];
    const float* rl   = (const float*)d_in[3];
    const float* xl   = (const float*)d_in[4];
    const float* Wq   = (const float*)d_in[5];
    const float* Wk   = (const float*)d_in[6];
    const float* Wv   = (const float*)d_in[7];
    const float* Wo   = (const float*)d_in[8];
    const float* gvw  = (const float*)d_in[9];
    const float* vemb = (const float*)d_in[10];
    const float* Wfc  = (const float*)d_in[11];
    const float* Wpr  = (const float*)d_in[12];

    constexpr int BT = 4096, T = 2048, D = 768, DFC = 3072, KV = 256;
    constexpr int VP = 32768, VOUT = 32000, L = 4;

    size_t off = 0;
    auto alloc = [&](size_t bytes) -> void* {
        void* p = (char*)d_ws + off;
        off += (bytes + 255) & ~(size_t)255;
        return p;
    };
    unsigned short* wqb  = (unsigned short*)alloc((size_t)L * D * D   * 2);
    unsigned short* wkb  = (unsigned short*)alloc((size_t)L * D * KV  * 2);
    unsigned short* wvb  = (unsigned short*)alloc((size_t)L * D * KV  * 2);
    unsigned short* wob  = (unsigned short*)alloc((size_t)L * D * D   * 2);
    unsigned short* wfcb = (unsigned short*)alloc((size_t)L * D * DFC * 2);
    unsigned short* wprb = (unsigned short*)alloc((size_t)L * DFC * D * 2);
    unsigned short* lmb  = (unsigned short*)alloc((size_t)D * VP * 2);
    float* x    = (float*)alloc((size_t)BT * D * 4);
    float* x0   = (float*)alloc((size_t)BT * D * 4);
    unsigned short* hb  = (unsigned short*)alloc((size_t)BT * D * 2);
    float* hf   = (float*)alloc((size_t)BT * D * 4);
    float* qb   = (float*)alloc((size_t)BT * D * 4);
    float* kb   = (float*)alloc((size_t)BT * KV * 4);
    float* vb   = (float*)alloc((size_t)BT * KV * 4);
    unsigned short* qbh = (unsigned short*)alloc((size_t)BT * D * 2);
    unsigned short* kbh = (unsigned short*)alloc((size_t)BT * KV * 2);
    unsigned short* vtb = (unsigned short*)alloc((size_t)BT * KV * 2);
    float* yb   = (float*)alloc((size_t)BT * D * 4);
    unsigned short* ybb  = (unsigned short*)alloc((size_t)BT * D * 2);
    float* fcb  = (float*)alloc((size_t)BT * DFC * 4);
    unsigned short* fcbb = (unsigned short*)alloc((size_t)BT * DFC * 2);

    auto cvt = [&](const float* s, unsigned short* dst, size_t n) {
        cvt_bf16_kernel<<<dim3(2048), dim3(256), 0, stream>>>(s, dst, n);
    };
    cvt(Wq,  wqb,  (size_t)L * D * D);
    cvt(Wk,  wkb,  (size_t)L * D * KV);
    cvt(Wv,  wvb,  (size_t)L * D * KV);
    cvt(Wo,  wob,  (size_t)L * D * D);
    cvt(Wfc, wfcb, (size_t)L * D * DFC);
    cvt(Wpr, wprb, (size_t)L * DFC * D);
    cvt(lmw, lmb,  (size_t)D * VP);

    auto gemm = [&](const unsigned short* A, const unsigned short* B, float* C,
                    const float* resid, int M, int N, int K, int ldc, int Nout, int ep) {
        dim3 grid(N / GBN, M / GBM);
        gemm_bf16_wmma<<<grid, dim3(256), 0, stream>>>(A, B, C, resid, M, N, K, ldc, Nout, ep);
    };

    embed_kernel<<<dim3(BT), dim3(256), 0, stream>>>(idx, wte, x, x0, D);

    const size_t nXD = (size_t)BT * D;
    for (int l = 0; l < L; ++l) {
        resid_kernel<<<dim3(2048), dim3(256), 0, stream>>>(x, x0, rl, xl, l, nXD);
        rmsnorm_out_kernel<<<dim3(BT), dim3(256), 0, stream>>>(x, hb, hf, D);

        gemm(hb, wqb + (size_t)l * D * D,  qb, nullptr, BT, D,  D, D,  D,  0);
        gemm(hb, wkb + (size_t)l * D * KV, kb, nullptr, BT, KV, D, KV, KV, 0);
        gemm(hb, wvb + (size_t)l * D * KV, vb, nullptr, BT, KV, D, KV, KV, 0);

        if (l == 1 || l == 3) {
            int j = (l == 1) ? 0 : 1;
            ve_kernel<<<dim3((BT * 4 + 255) / 256), dim3(256), 0, stream>>>(
                hf, idx, gvw, vemb, vb, j, BT);
        }

        rope_rms_kernel<<<dim3((BT * 12 + 7) / 8), dim3(256), 0, stream>>>(qb, qbh, BT, T, 12);
        rope_rms_kernel<<<dim3((BT * 4  + 7) / 8), dim3(256), 0, stream>>>(kb, kbh, BT, T, 4);
        vt_bf16_kernel<<<dim3(1024), dim3(256), 0, stream>>>(vb, vtb, T, BT);

        int W = (l == 3) ? 2048 : 1024;
        attn_wmma_kernel<<<dim3((BT / 16) * 12 / 8), dim3(256), 0, stream>>>(
            qbh, kbh, vtb, yb, T, BT, W);

        cvt(yb, ybb, nXD);
        gemm(ybb, wob + (size_t)l * D * D, x, x, BT, D, D, D, D, 1);

        rmsnorm_out_kernel<<<dim3(BT), dim3(256), 0, stream>>>(x, hb, nullptr, D);
        gemm(hb, wfcb + (size_t)l * D * DFC, fcb, nullptr, BT, DFC, D, DFC, DFC, 0);
        relu2_bf16_kernel<<<dim3(2048), dim3(256), 0, stream>>>(fcb, fcbb, (size_t)BT * DFC);
        gemm(fcbb, wprb + (size_t)l * DFC * D, x, x, BT, D, DFC, D, D, 1);
    }

    rmsnorm_out_kernel<<<dim3(BT), dim3(256), 0, stream>>>(x, hb, nullptr, D);
    gemm(hb, lmb, (float*)d_out, nullptr, BT, VP, D, VOUT, VOUT, 2);

    (void)in_sizes; (void)n_in; (void)out_size; (void)ws_size;
}
